// RGCN_22007412425073
// MI455X (gfx1250) — matrix operand order repaired and checked by reading, not compiled, not run on hardware
//
#include <hip/hip_runtime.h>
#include <hip/hip_bf16.h>

typedef __attribute__((ext_vector_type(16))) __bf16 v16bf;
typedef __attribute__((ext_vector_type(8)))  __bf16 v8bf;
typedef __attribute__((ext_vector_type(8)))  float  v8f;
typedef __attribute__((ext_vector_type(4)))  int    v4i;

static constexpr int FD = 128;   // feature dim (IN_F == H_F == 128)
static constexpr int RR = 4;     // relations
static constexpr int LDSP = FD + 8;  // padded LDS row stride (272B): 16 frag lanes hit distinct banks

__device__ __forceinline__ __bf16 f2bf(float f) { return (__bf16)f; }
__device__ __forceinline__ float  bf2f(__bf16 b) { return (float)b; }

// ---- async global->LDS staging (CDNA5 ASYNCcnt path), with guaranteed fallback ----
#if defined(__gfx1250__) && __has_builtin(__builtin_amdgcn_global_load_async_to_lds_b128) && \
    __has_builtin(__builtin_amdgcn_s_wait_asynccnt)
#define ASYNC_LDS 1
#else
#define ASYNC_LDS 0
#endif

#define AS_GLOBAL __attribute__((address_space(1)))
#define AS_SHARED __attribute__((address_space(3)))

__device__ __forceinline__ void stage16B(const __bf16* g, __bf16* l) {
#if ASYNC_LDS
  // builtin signature: (global int4*, shared int4*, imm offset, imm cpol)
  __builtin_amdgcn_global_load_async_to_lds_b128(
      (AS_GLOBAL v4i*)(v4i*)const_cast<__bf16*>(g),
      (AS_SHARED v4i*)(v4i*)l,
      0, 0);
#else
  *(v8bf*)l = *(const v8bf*)g;   // global_load_b128 + ds_store_b128
#endif
}

__device__ __forceinline__ void stage_wait() {
#if ASYNC_LDS
  __builtin_amdgcn_s_wait_asynccnt(0);
#endif
}

// ---------------- utility kernels ----------------
__global__ void k_zero_f32(float* __restrict__ p, int n) {
  int i = blockIdx.x * blockDim.x + threadIdx.x;
  if (i < n) p[i] = 0.0f;
}

__global__ void k_conv_bf16(const float* __restrict__ in, __bf16* __restrict__ out, int n) {
  int i = blockIdx.x * blockDim.x + threadIdx.x;
  if (i < n) out[i] = f2bf(in[i]);
}

// transpose-convert: in [K][Nc] row-major fp32 -> out [Nc][K] bf16 (B^T: contiguous K runs)
__global__ void k_convT_bf16(const float* __restrict__ in, __bf16* __restrict__ out, int K, int Nc) {
  int i = blockIdx.x * blockDim.x + threadIdx.x;
  if (i < K * Nc) {
    int k = i / Nc, c = i - k * Nc;
    out[(size_t)c * K + k] = f2bf(in[i]);
  }
}

__global__ void k_deg(const int* __restrict__ dst, float* __restrict__ deg, int total, int E, int N) {
  int i = blockIdx.x * blockDim.x + threadIdx.x;
  if (i < total) {
    int r = i / E;
    atomicAdd(deg + (size_t)r * N + dst[i], 1.0f);
  }
}

__global__ void k_rnorm(float* __restrict__ deg, int n) {
  int i = blockIdx.x * blockDim.x + threadIdx.x;
  if (i < n) deg[i] = 1.0f / fmaxf(deg[i], 1.0f);
}

// one block per edge, 128 threads = feature lanes; agg[dst] += X[src]  (L2-resident atomics)
__global__ void k_scatter_f32(const int* __restrict__ src, const int* __restrict__ dst,
                              const float* __restrict__ X, float* __restrict__ agg) {
  int e = blockIdx.x, f = threadIdx.x;
  int s = src[e], d = dst[e];
  atomicAdd(agg + (size_t)d * FD + f, X[(size_t)s * FD + f]);
}

__global__ void k_scatter_bf16(const int* __restrict__ src, const int* __restrict__ dst,
                               const __bf16* __restrict__ X, float* __restrict__ agg) {
  int e = blockIdx.x, f = threadIdx.x;
  int s = src[e], d = dst[e];
  atomicAdd(agg + (size_t)d * FD + f, bf2f(X[(size_t)s * FD + f]));
}

// aggb[n,f] = bf16(agg[n,f] * rnorm[n])   (one block per node)
__global__ void k_normconv(const float* __restrict__ agg, const float* __restrict__ rnorm,
                           __bf16* __restrict__ out) {
  int n = blockIdx.x, f = threadIdx.x;
  out[(size_t)n * FD + f] = f2bf(agg[(size_t)n * FD + f] * rnorm[n]);
}

__global__ void k_reluconv(const float* __restrict__ in, __bf16* __restrict__ out, int n) {
  int i = blockIdx.x * blockDim.x + threadIdx.x;
  if (i < n) out[i] = f2bf(fmaxf(in[i], 0.0f));
}

// ---------------- WMMA GEMM ----------------
// C[M,128] (+)= [A0 | A1][M, Kseg*128] @ B[Kseg*128, 128]  (+ bias, optional accumulate)
// A0/A1: bf16 row-major (lda=128). Bt: bf16 [col][Ktot] pre-transposed weights.
// Block = 128 threads = 4 waves; wave owns 32 rows x 128 cols (16 wmma accumulators).
// B is staged per 128-K segment into padded LDS (async DMA when available), shared by all waves;
// B fragments are software-pipelined one deep so ds_load latency overlaps the WMMA pair.
__global__ __launch_bounds__(128)
void k_gemm_wmma(const __bf16* __restrict__ A0, const __bf16* __restrict__ A1,
                 const __bf16* __restrict__ Bt, const float* __restrict__ bias,
                 float* __restrict__ C, int M, int Kseg, int accumulate) {
  __shared__ __bf16 Bs[FD * LDSP];   // 128 x (128+8) bf16 = 34 KB

  const int lane = threadIdx.x & 31;
  const int wave = threadIdx.x >> 5;
  const int rowBlock = blockIdx.x * 128 + wave * 32;
  const int m  = lane & 15;   // A row within tile / B,C column within tile
  const int hf = lane >> 4;   // lane half (K sub-range select / C row half)
  const int Ktot = Kseg * 128;

  v8f acc0[8] = {};
  v8f acc1[8] = {};

  int ar0 = rowBlock + m;      if (ar0 >= M) ar0 = M - 1;  // clamp: EXEC stays all-ones for WMMA
  int ar1 = rowBlock + 16 + m; if (ar1 >= M) ar1 = M - 1;

  // B fragment fetch from padded LDS: lanes 0-15 K=kl+0..7 & kl+16..23, lanes 16-31 K=kl+8..15 & kl+24..31
  auto ldB = [&](int ct, int kl) -> v16bf {
    v8bf lo = *(const v8bf*)&Bs[(ct * 16 + m) * LDSP + kl + hf * 8];
    v8bf hi = *(const v8bf*)&Bs[(ct * 16 + m) * LDSP + kl + 16 + hf * 8];
    return __builtin_shufflevector(lo, hi, 0,1,2,3,4,5,6,7,8,9,10,11,12,13,14,15);
  };

  for (int seg = 0; seg < Kseg; ++seg) {
    if (seg) __syncthreads();           // protect Bs before re-staging
    // cooperative stage of this 128x128 B segment: 2048 x 16B chunks over 128 threads
#pragma unroll
    for (int it = 0; it < 16; ++it) {
      int idx = threadIdx.x + it * 128;
      int col = idx >> 4;               // 16 chunks per column
      int kc  = (idx & 15) << 3;        // 8 bf16 per chunk
      stage16B(Bt + (size_t)col * Ktot + seg * 128 + kc, &Bs[col * LDSP + kc]);
    }
    stage_wait();
    __syncthreads();

    const __bf16* A = seg ? A1 : A0;
    const __bf16* ab0 = A + (size_t)ar0 * FD;
    const __bf16* ab1 = A + (size_t)ar1 * FD;

    v16bf bf_cur = ldB(0, 0);           // pipeline prime
#pragma unroll
    for (int kt = 0; kt < 4; ++kt) {
      const int kl = kt * 32;
      // 16-bit A 16x32 layout: lanes 0-15 K=0..7 & 16..23, lanes 16-31 K=8..15 & 24..31
      v8bf a0lo = *(const v8bf*)(ab0 + kl + hf * 8);
      v8bf a0hi = *(const v8bf*)(ab0 + kl + 16 + hf * 8);
      v8bf a1lo = *(const v8bf*)(ab1 + kl + hf * 8);
      v8bf a1hi = *(const v8bf*)(ab1 + kl + 16 + hf * 8);
      v16bf af0 = __builtin_shufflevector(a0lo, a0hi, 0,1,2,3,4,5,6,7,8,9,10,11,12,13,14,15);
      v16bf af1 = __builtin_shufflevector(a1lo, a1hi, 0,1,2,3,4,5,6,7,8,9,10,11,12,13,14,15);
      __builtin_prefetch(ab0 + kl + 64, 0, 1);
#pragma unroll
      for (int ct = 0; ct < 8; ++ct) {
        v16bf bf_use = bf_cur;
        // issue next fragment's ds_load before consuming the current one
        if (!(kt == 3 && ct == 7)) {
          int nct = (ct == 7) ? 0 : ct + 1;
          int nkl = (ct == 7) ? kl + 32 : kl;
          bf_cur = ldB(nct, nkl);
        }
        acc0[ct] = __builtin_amdgcn_wmma_f32_16x16x32_bf16(
            false, af0, false, bf_use, (short)0, acc0[ct], false, false);
        acc1[ct] = __builtin_amdgcn_wmma_f32_16x16x32_bf16(
            false, af1, false, bf_use, (short)0, acc1[ct], false, false);
      }
    }
  }

  // epilogue: C layout lane m = col within tile, VGPR i -> row (lane>>4)*8 + i
#pragma unroll
  for (int ct = 0; ct < 8; ++ct) {
    const int col = ct * 16 + m;
    const float badd = bias ? bias[col] : 0.0f;
#pragma unroll
    for (int i = 0; i < 8; ++i) {
      int r0 = rowBlock + hf * 8 + i;
      int r1 = r0 + 16;
      if (r0 < M) {
        float v = acc0[ct][i] + badd;
        float* cp = C + (size_t)r0 * FD + col;
        if (accumulate) v += *cp;
        *cp = v;
      }
      if (r1 < M) {
        float v = acc1[ct][i] + badd;
        float* cp = C + (size_t)r1 * FD + col;
        if (accumulate) v += *cp;
        *cp = v;
      }
    }
  }
}

// logits[n,c] = bo2[c] + sum_k t[n,k] * Wo2[k,c]   (C=2)
__global__ void k_logits(const __bf16* __restrict__ t, const float* __restrict__ Wo2,
                         const float* __restrict__ bo2, float* __restrict__ out, int N) {
  int n = blockIdx.x * blockDim.x + threadIdx.x;
  if (n < N) {
    float a0 = bo2[0], a1 = bo2[1];
    const __bf16* tp = t + (size_t)n * FD;
#pragma unroll 4
    for (int k = 0; k < FD; ++k) {
      float tv = bf2f(tp[k]);
      a0 += tv * Wo2[2 * k + 0];
      a1 += tv * Wo2[2 * k + 1];
    }
    out[2 * n + 0] = a0;
    out[2 * n + 1] = a1;
  }
}

// ---------------- host-side orchestration ----------------
extern "C" void kernel_launch(void* const* d_in, const int* in_sizes, int n_in,
                              void* d_out, int out_size, void* d_ws, size_t ws_size,
                              hipStream_t stream) {
  const float* x     = (const float*)d_in[0];
  const int*   src   = (const int*)d_in[1];
  const int*   dst   = (const int*)d_in[2];
  const float* W1    = (const float*)d_in[3];
  const float* loop1 = (const float*)d_in[4];
  const float* b1    = (const float*)d_in[5];
  const float* W2    = (const float*)d_in[6];
  const float* loop2 = (const float*)d_in[7];
  const float* b2    = (const float*)d_in[8];
  const float* Wo1   = (const float*)d_in[9];
  const float* bo1   = (const float*)d_in[10];
  const float* Wo2   = (const float*)d_in[11];
  const float* bo2   = (const float*)d_in[12];
  float* out = (float*)d_out;

  const int N = in_sizes[0] / FD;        // 100000
  const int E = in_sizes[1] / RR;        // 160000

  // workspace carve-out (256B aligned slices)
  size_t off = 0;
  char* base = (char*)d_ws;
  auto balloc = [&](size_t bytes) -> char* {
    char* p = base + off;
    off += (bytes + 255) & ~(size_t)255;
    return p;
  };
  float*  rnorm = (float*) balloc((size_t)RR * N * 4);
  __bf16* xb    = (__bf16*)balloc((size_t)N * FD * 2);
  __bf16* hbA   = (__bf16*)balloc((size_t)N * FD * 2);
  __bf16* hbB   = (__bf16*)balloc((size_t)N * FD * 2);
  float*  hacc  = (float*) balloc((size_t)N * FD * 4);
  float*  agg   = (float*) balloc((size_t)N * FD * 4);
  __bf16* aggb  = (__bf16*)balloc((size_t)N * FD * 2);   // reused as t-buffer at the end
  __bf16* wL1   = (__bf16*)balloc((size_t)FD * FD * 2);          // loop1^T
  __bf16* wR1   = (__bf16*)balloc((size_t)RR * FD * FD * 2);     // W1[r]^T
  __bf16* wL2   = (__bf16*)balloc((size_t)FD * FD * 2);
  __bf16* wR2   = (__bf16*)balloc((size_t)RR * FD * FD * 2);
  __bf16* wO1   = (__bf16*)balloc((size_t)2 * FD * FD * 2);      // Wo1^T [128][256]

  const int TB = 256;
  const int nNF = N * FD;
  dim3 gGemm((N + 127) / 128), bGemm(128);

  // degrees -> rnorm
  k_zero_f32<<<(RR * N + TB - 1) / TB, TB, 0, stream>>>(rnorm, RR * N);
  k_deg<<<(RR * E + TB - 1) / TB, TB, 0, stream>>>(dst, rnorm, RR * E, E, N);
  k_rnorm<<<(RR * N + TB - 1) / TB, TB, 0, stream>>>(rnorm, RR * N);

  // precision conversion / weight transposes
  k_conv_bf16<<<(nNF + TB - 1) / TB, TB, 0, stream>>>(x, xb, nNF);
  k_convT_bf16<<<(FD * FD + TB - 1) / TB, TB, 0, stream>>>(loop1, wL1, FD, FD);
  k_convT_bf16<<<(FD * FD + TB - 1) / TB, TB, 0, stream>>>(loop2, wL2, FD, FD);
  for (int r = 0; r < RR; ++r) {
    k_convT_bf16<<<(FD * FD + TB - 1) / TB, TB, 0, stream>>>(W1 + (size_t)r * FD * FD, wR1 + (size_t)r * FD * FD, FD, FD);
    k_convT_bf16<<<(FD * FD + TB - 1) / TB, TB, 0, stream>>>(W2 + (size_t)r * FD * FD, wR2 + (size_t)r * FD * FD, FD, FD);
  }
  k_convT_bf16<<<(2 * FD * FD + TB - 1) / TB, TB, 0, stream>>>(Wo1, wO1, 2 * FD, FD);

  // ---- layer 1 ----
  k_gemm_wmma<<<gGemm, bGemm, 0, stream>>>(xb, nullptr, wL1, b1, hacc, N, 1, 0);
  for (int r = 0; r < RR; ++r) {
    k_zero_f32<<<(nNF + TB - 1) / TB, TB, 0, stream>>>(agg, nNF);
    k_scatter_f32<<<E, FD, 0, stream>>>(src + (size_t)r * E, dst + (size_t)r * E, x, agg);
    k_normconv<<<N, FD, 0, stream>>>(agg, rnorm + (size_t)r * N, aggb);
    k_gemm_wmma<<<gGemm, bGemm, 0, stream>>>(aggb, nullptr, wR1 + (size_t)r * FD * FD, nullptr, hacc, N, 1, 1);
  }
  k_reluconv<<<(nNF + TB - 1) / TB, TB, 0, stream>>>(hacc, hbA, nNF);

  // ---- layer 2 ----
  k_gemm_wmma<<<gGemm, bGemm, 0, stream>>>(hbA, nullptr, wL2, b2, hacc, N, 1, 0);
  for (int r = 0; r < RR; ++r) {
    k_zero_f32<<<(nNF + TB - 1) / TB, TB, 0, stream>>>(agg, nNF);
    k_scatter_bf16<<<E, FD, 0, stream>>>(src + (size_t)r * E, dst + (size_t)r * E, hbA, agg);
    k_normconv<<<N, FD, 0, stream>>>(agg, rnorm + (size_t)r * N, aggb);
    k_gemm_wmma<<<gGemm, bGemm, 0, stream>>>(aggb, nullptr, wR2 + (size_t)r * FD * FD, nullptr, hacc, N, 1, 1);
  }
  k_reluconv<<<(nNF + TB - 1) / TB, TB, 0, stream>>>(hacc, hbB, nNF);

  // ---- output MLP: relu([h2 | x] @ Wo1 + bo1) @ Wo2 + bo2 ----
  k_gemm_wmma<<<gGemm, bGemm, 0, stream>>>(hbB, xb, wO1, bo1, hacc, N, 2, 0);
  k_reluconv<<<(nNF + TB - 1) / TB, TB, 0, stream>>>(hacc, aggb, nNF);   // t in bf16
  k_logits<<<(N + TB - 1) / TB, TB, 0, stream>>>(aggb, Wo2, bo2, out, N);
}
